// ScaledDotProductAttention_43714177138943
// MI455X (gfx1250) — matrix operand order repaired and checked by reading, not compiled, not run on hardware
//
#include <hip/hip_runtime.h>

// Problem constants (match reference)
#define B_ 4
#define H_ 16
#define S_ 2048
#define D_ 64

typedef __attribute__((ext_vector_type(16))) _Float16 v16h;
typedef __attribute__((ext_vector_type(8)))  float    v8f;
typedef __attribute__((ext_vector_type(4)))  float    f4;   // native vector (nontemporal-store friendly)
typedef __attribute__((ext_vector_type(4)))  _Float16 h4;

constexpr int QTILE   = 16;          // q rows per workgroup
constexpr int KBLK    = 128;         // keys staged per iteration (8 waves x 16)
constexpr int THREADS = 256;         // 8 wave32
constexpr int SC_STRIDE = S_ + 16;   // padded row stride (floats); 2064*4 B = 16B-aligned rows

// Dynamic LDS layout
constexpr size_t SC_BYTES  = (size_t)QTILE * SC_STRIDE * sizeof(float);  // 132096: score/prob rows
constexpr size_t KST_BYTES = (size_t)KBLK * D_ * sizeof(_Float16);       //  16384: K/V staging (f16)
constexpr size_t QT_BYTES  = (size_t)QTILE * D_ * sizeof(_Float16);      //   2048: Q tile (f16, pre-scaled)
constexpr size_t RED_BYTES = (size_t)4 * 256 * sizeof(float);            //   4096: cross-wave PV reduce
constexpr size_t INV_BYTES = (size_t)QTILE * sizeof(float);              //     64: per-row 1/sum
constexpr size_t SMEM_BYTES = SC_BYTES + KST_BYTES + QT_BYTES + RED_BYTES + INV_BYTES;

__global__ __launch_bounds__(THREADS)
void attn_fwd_kernel(const float* __restrict__ Q, const float* __restrict__ K,
                     const float* __restrict__ V, float* __restrict__ Out,
                     float* __restrict__ Attn)
{
    extern __shared__ char smem[];
    float*    sc     = (float*)smem;                                   // [16][SC_STRIDE]
    _Float16* kst    = (_Float16*)(smem + SC_BYTES);                   // [128][64]
    _Float16* qt     = (_Float16*)(smem + SC_BYTES + KST_BYTES);       // [16][64]
    float*    red    = (float*)(smem + SC_BYTES + KST_BYTES + QT_BYTES);
    float*    invrow = (float*)(smem + SC_BYTES + KST_BYTES + QT_BYTES + RED_BYTES);

    const int tid  = threadIdx.x;
    const int w    = tid >> 5;        // wave id 0..7
    const int lane = tid & 31;
    const int li   = lane & 15;       // position within 16-lane half
    const int hi   = lane >> 4;       // 0 = lanes 0-15, 1 = lanes 16-31

    const int qb   = blockIdx.x & (S_ / QTILE - 1);   // 128 q-tiles per (b,h)
    const int bh   = blockIdx.x >> 7;
    const int q0   = qb * QTILE;
    const int kmax = q0 + QTILE;      // causal: keys [0, kmax) can be nonzero

    const float* qp = Q + (size_t)bh * S_ * D_;
    const float* kp = K + (size_t)bh * S_ * D_;
    const float* vp = V + (size_t)bh * S_ * D_;

    // ---- Stage Q tile: f32 -> f16, pre-divided by temperature (sqrt(64)=8) ----
    {
        const f4* q4 = (const f4*)(qp + (size_t)q0 * D_);
        for (int i = tid; i < QTILE * D_ / 4; i += THREADS) {
            f4 f = q4[i];
            h4 h; h.x = (_Float16)(f.x * 0.125f); h.y = (_Float16)(f.y * 0.125f);
                  h.z = (_Float16)(f.z * 0.125f); h.w = (_Float16)(f.w * 0.125f);
            *(h4*)(qt + i * 4) = h;               // ds_store_b64
        }
    }

    // ---- Phase 1: scores S = (Q/8) K^T with causal mask, full rows into LDS ----
    for (int base = 0; base < kmax; base += KBLK) {
        const int nk = min(KBLK, kmax - base);    // multiple of 16
        __syncthreads();                          // kst reuse / qt ready
        {
            const f4* k4 = (const f4*)(kp + (size_t)base * D_);
            for (int i = tid; i < nk * D_ / 4; i += THREADS) {
                f4 f = k4[i];
                h4 h; h.x = (_Float16)f.x; h.y = (_Float16)f.y;
                      h.z = (_Float16)f.z; h.w = (_Float16)f.w;
                *(h4*)(kst + i * 4) = h;
            }
        }
        __syncthreads();
        if (w * 16 < nk) {                        // wave-uniform: EXEC all ones at WMMA
            const int ktl = w * 16;               // local key-tile start
            v8f c = {};
            #pragma unroll
            for (int db = 0; db < D_; db += 32) { // K-dim (d) = 64 -> 2 WMMAs
                v16h a, b;
                #pragma unroll
                for (int h = 0; h < 16; ++h) {
                    const int ka = (h < 8 ? h : h + 8) + (hi ? 8 : 0);   // A K-index 0..31
                    const int kb = (h < 8 ? h : h + 8) + (hi ? 8 : 0);   // B K-index 0..31
                    a[h] = qt[li * D_ + db + ka];                        // A row = li
                    b[h] = kst[(ktl + li) * D_ + db + kb];               // B col(key) = li
                }
                c = __builtin_amdgcn_wmma_f32_16x16x32_f16(false, a, false, b,
                                                           (short)0, c, false, false);
            }
            const int keyn = base + ktl + li;     // this lane's key column
            #pragma unroll
            for (int j = 0; j < 8; ++j) {
                const int m = j + (hi ? 8 : 0);   // D row
                sc[m * SC_STRIDE + keyn] = (keyn <= q0 + m) ? c[j] : -1000000000.0f;
            }
        }
    }
    __syncthreads();

    // ---- Phase 2: row max + exp (unnormalized) + sum; keep 1/sum per row ----
    {
        const int row = 2 * w + hi;               // 8 waves x 2 halves = 16 rows
        float* srow = sc + row * SC_STRIDE;
        float rmax = -3.4e38f;
        for (int c = li * 4; c < kmax; c += 64) { // vectorized: ds_load_b128
            f4 v = *(const f4*)(srow + c);
            rmax = fmaxf(rmax, fmaxf(fmaxf(v.x, v.y), fmaxf(v.z, v.w)));
        }
        #pragma unroll
        for (int off = 8; off; off >>= 1) rmax = fmaxf(rmax, __shfl_xor(rmax, off, 16));
        float rsum = 0.0f;
        for (int c = li * 4; c < kmax; c += 64) { // single fused exp+sum pass, in-place
            f4 v = *(const f4*)(srow + c);
            f4 e; e.x = expf(v.x - rmax); e.y = expf(v.y - rmax);
                  e.z = expf(v.z - rmax); e.w = expf(v.w - rmax);
            *(f4*)(srow + c) = e;
            rsum += (e.x + e.y) + (e.z + e.w);
        }
        #pragma unroll
        for (int off = 8; off; off >>= 1) rsum += __shfl_xor(rsum, off, 16);
        if (li == 0) invrow[row] = 1.0f / rsum;   // normalization deferred to consumers
    }
    __syncthreads();

    // ---- Phase 3: stream normalized prob tile to global attn (128-bit NT stores) ----
    {
        float* arow0 = Attn + ((size_t)bh * S_ + q0) * S_;
        for (int i = tid; i < QTILE * S_ / 4; i += THREADS) {
            const int r = i >> 9;                 // 512 float4 per row
            const int c = (i & 511) << 2;
            f4 pv;
            if (c < kmax) {                       // kmax % 16 == 0: no straddle
                const float inv = invrow[r];
                f4 e = *(const f4*)(sc + r * SC_STRIDE + c);
                pv.x = e.x * inv; pv.y = e.y * inv; pv.z = e.z * inv; pv.w = e.w * inv;
            } else {
                pv = (f4){0.0f, 0.0f, 0.0f, 0.0f};
            }
            __builtin_nontemporal_store(pv, (f4*)(arow0 + (size_t)r * S_ + c));
        }
    }

    // ---- Phase 4: Out = (exp P) @ V, scale by 1/sum at the end ----
    const int dt = w & 3;                         // d-column tile (16 wide)
    const int kh = (w >> 2) * 64;                 // key half within a 128-block
    const int dc = dt * 16;
    v8f acc = {};
    for (int base = 0; base < kmax; base += KBLK) {
        __syncthreads();                          // vstage (kst) reuse
        {
            const f4* v4 = (const f4*)(vp + (size_t)base * D_);
            for (int i = tid; i < KBLK * D_ / 4; i += THREADS) {   // base+127 <= 2047 always
                f4 f = v4[i];
                h4 h; h.x = (_Float16)f.x; h.y = (_Float16)f.y;
                      h.z = (_Float16)f.z; h.w = (_Float16)f.w;
                *(h4*)(kst + i * 4) = h;
            }
        }
        __syncthreads();
        #pragma unroll
        for (int s32 = 0; s32 < 64; s32 += 32) {
            const int kk   = kh + s32;            // local key base in staged block
            const int keyg = base + kk;           // global key base (wave-uniform)
            if (keyg < kmax) {
                v16h a, b;
                #pragma unroll
                for (int h = 0; h < 16; ++h) {
                    const int ka  = (h < 8 ? h : h + 8) + (hi ? 8 : 0);
                    const int kb  = (h < 8 ? h : h + 8) + (hi ? 8 : 0);
                    const int key = keyg + ka;
                    a[h] = (key < kmax) ? (_Float16)sc[li * SC_STRIDE + key] : (_Float16)0.0f;
                    b[h] = kst[(kk + kb) * D_ + dc + li];
                }
                acc = __builtin_amdgcn_wmma_f32_16x16x32_f16(false, a, false, b,
                                                             (short)0, acc, false, false);
            }
        }
    }

    // ---- Cross-wave pair reduction + normalized output store ----
    __syncthreads();
    if (w >= 4) {
        #pragma unroll
        for (int j = 0; j < 8; ++j) red[(w - 4) * 256 + lane * 8 + j] = acc[j];
    }
    __syncthreads();
    if (w < 4) {
        float* orow = Out + ((size_t)bh * S_ + q0) * D_;
        #pragma unroll
        for (int j = 0; j < 8; ++j) {
            const int m = j + (hi ? 8 : 0);
            const float s = (acc[j] + red[w * 256 + lane * 8 + j]) * invrow[m];
            __builtin_nontemporal_store(s, orow + (size_t)m * D_ + dc + li);
        }
    }
}

extern "C" void kernel_launch(void* const* d_in, const int* in_sizes, int n_in,
                              void* d_out, int out_size, void* d_ws, size_t ws_size,
                              hipStream_t stream) {
    (void)in_sizes; (void)n_in; (void)out_size; (void)d_ws; (void)ws_size;
    const float* q = (const float*)d_in[0];
    const float* k = (const float*)d_in[1];
    const float* v = (const float*)d_in[2];
    // d_in[3] is the causal mask; causality is applied analytically in-kernel.
    float* out  = (float*)d_out;                                   // [B,H,S,D]
    float* attn = out + (size_t)B_ * H_ * S_ * D_;                 // [B,H,S,S]

    dim3 grid(B_ * H_ * (S_ / QTILE));   // 8192 workgroups
    attn_fwd_kernel<<<grid, THREADS, SMEM_BYTES, stream>>>(q, k, v, out, attn);
}
